// GCN_edge_prev_27908697489551
// MI455X (gfx1250) — matrix operand order repaired and checked by reading, not compiled, not run on hardware
//
#include <hip/hip_runtime.h>
#include <hip/hip_bf16.h>

typedef __attribute__((ext_vector_type(16))) __bf16 v16bf;
typedef __attribute__((ext_vector_type(8)))  __bf16 v8bf;
typedef __attribute__((ext_vector_type(8)))  float  v8f;
typedef __attribute__((ext_vector_type(4)))  unsigned int v4u;
typedef __attribute__((ext_vector_type(8)))  int v8i;
typedef __attribute__((ext_vector_type(4)))  int v4i;

#define N_CH 128
#define TS   136          // LDS tile row stride (bf16): 272B = 256B row + 16B TDM pad
#define ROWS_PB 64        // rows per block (4 row-tiles of 16)
#define BN_EPS 1e-5f

// ---------------- helpers ----------------
__device__ __forceinline__ void fadd(float* p, float v) {
    unsafeAtomicAdd(p, v);   // hardware global_atomic_add_f32
}

// ---------------- gcn_norm ----------------
__global__ void k_init_deg(float* __restrict__ deg, int n) {
    int i = blockIdx.x * blockDim.x + threadIdx.x;
    if (i < n) deg[i] = 1.0f;   // self-loop weight
}

__global__ void k_deg_edges(const int* __restrict__ col, const float* __restrict__ w,
                            float* __restrict__ deg, int e) {
    int i = blockIdx.x * blockDim.x + threadIdx.x;
    if (i < e) fadd(&deg[col[i]], w[i]);
}

__global__ void k_dinv(const float* __restrict__ deg, float* __restrict__ dinv, int n) {
    int i = blockIdx.x * blockDim.x + threadIdx.x;
    if (i < n) {
        float d = deg[i];
        dinv[i] = (d > 0.0f) ? rsqrtf(fmaxf(d, 1e-30f)) : 0.0f;
    }
}

__global__ void k_norm(const int* __restrict__ row, const int* __restrict__ col,
                       const float* __restrict__ w, const float* __restrict__ dinv,
                       float* __restrict__ nrm, int e) {
    int i = blockIdx.x * blockDim.x + threadIdx.x;
    if (i < e) nrm[i] = dinv[row[i]] * w[i] * dinv[col[i]];
}

// ---------------- f32 -> bf16 conversions ----------------
__global__ void k_f32_to_bf16(const float* __restrict__ src, __bf16* __restrict__ dst, long long n) {
    long long i = blockIdx.x * (long long)blockDim.x + threadIdx.x;
    if (i < n) dst[i] = (__bf16)src[i];
}

// weights: emit transposed bf16  Wt[n][k] = W[k][n]  (so B fragments are lane-contiguous)
__global__ void k_w_to_bf16t(const float* __restrict__ w, __bf16* __restrict__ wt, int n) {
    int i = blockIdx.x * blockDim.x + threadIdx.x;
    if (i < n) {
        int nn = i >> 7, kk = i & 127;
        wt[i] = (__bf16)w[kk * N_CH + nn];
    }
}

// ---------------- WMMA GEMM: C[n x 128] = A[n x 128](bf16) @ W[128 x 128] ----------------
// 64-row A tile staged to LDS by the Tensor Data Mover; B preloaded once per wave from
// transposed bf16 W and reused across the 4 row-tiles.
__global__ __launch_bounds__(256) void k_gemm_bf16(
        const __bf16* __restrict__ A, const __bf16* __restrict__ Wt,
        float* __restrict__ C, int nrows) {
    __shared__ __bf16 tA[ROWS_PB * TS];
    const int row0 = blockIdx.x * ROWS_PB;
    const int t = threadIdx.x;

    // ---- TDM: DMA 64 x 128 bf16 tile -> LDS, 16B pad per 256B row (row stride = TS) ----
    if (t < 32) {
        unsigned ldsa = (unsigned)(size_t)(void*)tA;
        unsigned long long ga = (unsigned long long)(const void*)(A + (size_t)row0 * N_CH);
        unsigned rem = (unsigned)(nrows - row0);     // remaining rows; OOB rows read 0
        v4u g0;
        g0[0] = 1u;                                          // count=1, user descriptor
        g0[1] = ldsa;                                        // lds_addr
        g0[2] = (unsigned)(ga & 0xFFFFFFFFull);              // global_addr[31:0]
        g0[3] = (unsigned)((ga >> 32) & 0x1FFFFFFull)        // global_addr[56:32]
                | (2u << 30);                                // type=2 (image)
        v8i g1;
        g1[0] = (int)((1u << 16)      // data_size=1 -> 2 bytes
                    | (1u << 20)      // pad_enable
                    | (5u << 22)      // pad_interval: 64 DWORDs (one 256B row)
                    | (3u << 25));    // pad_amount: 4 DWORDs (16B)
        g1[1] = (int)(128u << 16);                           // tensor_dim0 = 128
        g1[2] = (int)((rem & 0xFFFFu) << 16);                // tensor_dim1[15:0]
        g1[3] = (int)((rem >> 16) | (128u << 16));           // tensor_dim1[31:16] | tile_dim0=128
        g1[4] = ROWS_PB;                                     // tile_dim1 = 64 rows
        g1[5] = 128;                                         // tensor_dim0_stride = 128
        g1[6] = 0;
        g1[7] = 0;
        v4i z4 = {};
        v8i z8 = {};
        __builtin_amdgcn_tensor_load_to_lds(g0, g1, z4, z4, z8, 0);
        __builtin_amdgcn_s_wait_tensorcnt(0);
    }
    __syncthreads();

    const int lane = t & 31, wave = t >> 5;
    const int half = lane >> 4, mr = lane & 15;
    const int n0 = wave * 16;                   // this wave's 16-column slab

    // preload B fragments for all 4 K-steps (reused across row-tiles)
    const __bf16* brow = Wt + (size_t)(n0 + mr) * N_CH;
    v16bf b[4];
    #pragma unroll
    for (int ks = 0; ks < 4; ++ks) {
        v8bf blo = *(const v8bf*)(brow + ks * 32 + 8 * half);
        v8bf bhi = *(const v8bf*)(brow + ks * 32 + 16 + 8 * half);
        b[ks] = __builtin_shufflevector(blo, bhi,
                    0, 1, 2, 3, 4, 5, 6, 7, 8, 9, 10, 11, 12, 13, 14, 15);
    }

    #pragma unroll
    for (int rt = 0; rt < 4; ++rt) {
        const __bf16* arow = &tA[(rt * 16 + mr) * TS];
        v8f acc = {};
        #pragma unroll
        for (int ks = 0; ks < 4; ++ks) {        // K = 128 in steps of 32
            // A 16x32 bf16 layout: per lane two contiguous 16B runs
            v8bf alo = *(const v8bf*)(arow + ks * 32 + 8 * half);
            v8bf ahi = *(const v8bf*)(arow + ks * 32 + 16 + 8 * half);
            v16bf a = __builtin_shufflevector(alo, ahi,
                          0, 1, 2, 3, 4, 5, 6, 7, 8, 9, 10, 11, 12, 13, 14, 15);
            acc = __builtin_amdgcn_wmma_f32_16x16x32_bf16(false, a, false, b[ks],
                                                          (short)0, acc, false, false);
        }
        // C/D: VGPR r -> (M=r, N=lane) lanes 0-15 ; (M=r+8, N=lane-16) lanes 16-31
        const int rbase = row0 + rt * 16;
        float* cptr = C + (size_t)(rbase + 8 * half) * N_CH + n0 + mr;
        if (rbase + 16 <= nrows) {              // uniform fast path: no per-lane guards
            #pragma unroll
            for (int r = 0; r < 8; ++r)
                cptr[(size_t)r * N_CH] = acc[r];
        } else {
            #pragma unroll
            for (int r = 0; r < 8; ++r) {
                const int row = rbase + r + 8 * half;
                if (row < nrows) C[(size_t)row * N_CH + n0 + mr] = acc[r];
            }
        }
    }
}

// ---------------- aggregation (self loop init + edge scatter) ----------------
__global__ void k_agg_init(const float* __restrict__ xw, const float* __restrict__ dinv,
                           float* __restrict__ h, long long n128) {
    long long i = blockIdx.x * (long long)blockDim.x + threadIdx.x;
    if (i < n128) {
        int node = (int)(i >> 7);
        float dv = dinv[node];
        h[i] = dv * dv * xw[i];
    }
}

__global__ void k_agg_edges(const int* __restrict__ row, const int* __restrict__ col,
                            const float* __restrict__ nrm, const float* __restrict__ xw,
                            float* __restrict__ h, int e) {
    long long tid = blockIdx.x * (long long)blockDim.x + threadIdx.x;
    if (tid >= (long long)e * 32) return;
    int  ei = (int)(tid >> 5);
    int  c4 = ((int)tid & 31) * 4;
    int  r  = row[ei], cc = col[ei];
    float nw = nrm[ei];
    const float4 v = *(const float4*)(xw + (size_t)r * N_CH + c4);  // L2-resident gather
    float* dst = h + (size_t)cc * N_CH + c4;
    fadd(dst + 0, nw * v.x);
    fadd(dst + 1, nw * v.y);
    fadd(dst + 2, nw * v.z);
    fadd(dst + 3, nw * v.w);
}

// ---------------- BatchNorm ----------------
__global__ void k_zero_stats(float* __restrict__ s) {
    int i = threadIdx.x;
    if (i < 256) s[i] = 0.0f;
}

__global__ void k_bn_stats(const float* __restrict__ h, const float* __restrict__ bias,
                           float* __restrict__ stats, int n) {
    const int c = threadIdx.x;           // blockDim = 128
    const float bc = bias[c];
    float s = 0.0f, s2 = 0.0f;
    for (int r = blockIdx.x; r < n; r += gridDim.x) {
        float v = h[(size_t)r * N_CH + c] + bc;
        s += v; s2 += v * v;
    }
    fadd(&stats[c], s);
    fadd(&stats[128 + c], s2);
}

// apply BN + ReLU, emit bf16 input for the next WMMA GEMM
__global__ void k_bn_apply(const float* __restrict__ h, const float* __restrict__ bias,
                           const float* __restrict__ gamma, const float* __restrict__ beta,
                           const float* __restrict__ stats, __bf16* __restrict__ xb,
                           int n, long long n128) {
    long long i = blockIdx.x * (long long)blockDim.x + threadIdx.x;
    if (i >= n128) return;
    int c = (int)(i & 127);
    float invn = 1.0f / (float)n;
    float mean = stats[c] * invn;
    float var  = stats[128 + c] * invn - mean * mean;
    float k    = rsqrtf(var + BN_EPS) * gamma[c];
    float v    = (h[i] + bias[c] - mean) * k + beta[c];
    xb[i] = (__bf16)fmaxf(v, 0.0f);
}

// ---------------- layer 3: 128 -> 1 ----------------
__global__ void k_matvec(const __bf16* __restrict__ xb, const float* __restrict__ w3,
                         float* __restrict__ y, int n) {
    int i = blockIdx.x * blockDim.x + threadIdx.x;
    if (i >= n) return;
    const __bf16* r = xb + (size_t)i * N_CH;
    float acc = 0.0f;
    #pragma unroll 8
    for (int c = 0; c < N_CH; ++c) acc += (float)r[c] * w3[c];
    y[i] = acc;
}

__global__ void k_out_init(const float* __restrict__ xw3, const float* __restrict__ dinv,
                           const float* __restrict__ b3, float* __restrict__ out, int n) {
    int i = blockIdx.x * blockDim.x + threadIdx.x;
    if (i < n) {
        float dv = dinv[i];
        out[i] = b3[0] + dv * dv * xw3[i];
    }
}

__global__ void k_out_edges(const int* __restrict__ row, const int* __restrict__ col,
                            const float* __restrict__ nrm, const float* __restrict__ xw3,
                            float* __restrict__ out, int e) {
    int i = blockIdx.x * blockDim.x + threadIdx.x;
    if (i < e) fadd(&out[col[i]], nrm[i] * xw3[row[i]]);
}

// ---------------- driver ----------------
extern "C" void kernel_launch(void* const* d_in, const int* in_sizes, int n_in,
                              void* d_out, int out_size, void* d_ws, size_t ws_size,
                              hipStream_t stream) {
    const float* x   = (const float*)d_in[0];
    const int*   ei  = (const int*)d_in[1];
    const float* ew  = (const float*)d_in[2];
    const float* W1  = (const float*)d_in[3];
    const float* b1  = (const float*)d_in[4];
    const float* g1  = (const float*)d_in[5];
    const float* be1 = (const float*)d_in[6];
    const float* W2  = (const float*)d_in[7];
    const float* b2  = (const float*)d_in[8];
    const float* g2  = (const float*)d_in[9];
    const float* be2 = (const float*)d_in[10];
    const float* W3  = (const float*)d_in[11];
    const float* b3  = (const float*)d_in[12];
    float* out = (float*)d_out;

    const int N = in_sizes[0] / N_CH;
    const int E = in_sizes[2];
    const int* row = ei;        // edge_index[0] = source
    const int* col = ei + E;    // edge_index[1] = target
    const long long n128 = (long long)N * N_CH;

    // workspace carve-up (256B aligned slots)
    char* wsp = (char*)d_ws;
    size_t off = 0;
    auto carve = [&](size_t bytes) -> char* {
        char* p = wsp + off;
        off += (bytes + 255) & ~(size_t)255;
        return p;
    };
    float*  deg   = (float*) carve((size_t)N * 4);
    float*  dinv  = (float*) carve((size_t)N * 4);
    float*  nrm   = (float*) carve((size_t)E * 4);
    float*  xw    = (float*) carve((size_t)n128 * 4);   // GEMM out / layer3 scalar reuse
    float*  h     = (float*) carve((size_t)n128 * 4);   // aggregation out
    __bf16* xb    = (__bf16*)carve((size_t)n128 * 2);   // bf16 GEMM input
    __bf16* wt    = (__bf16*)carve((size_t)N_CH * N_CH * 2); // transposed bf16 weights
    float*  stats = (float*) carve(256 * 4);

    const int B = 256;
    const int gN  = (N + B - 1) / B;
    const int gE  = (E + B - 1) / B;
    const int gNC = (int)((n128 + B - 1) / B);
    const int gEC = (int)(((long long)E * 32 + B - 1) / B);
    const int gT  = (N + ROWS_PB - 1) / ROWS_PB;
    const int gW  = (N_CH * N_CH + B - 1) / B;

    // gcn_norm
    k_init_deg<<<gN, B, 0, stream>>>(deg, N);
    k_deg_edges<<<gE, B, 0, stream>>>(col, ew, deg, E);
    k_dinv<<<gN, B, 0, stream>>>(deg, dinv, N);
    k_norm<<<gE, B, 0, stream>>>(row, col, ew, dinv, nrm, E);

    // ---- layer 1 ----
    k_f32_to_bf16<<<gNC, B, 0, stream>>>(x, xb, n128);
    k_w_to_bf16t<<<gW, B, 0, stream>>>(W1, wt, N_CH * N_CH);
    k_gemm_bf16<<<gT, B, 0, stream>>>(xb, wt, xw, N);
    k_agg_init<<<gNC, B, 0, stream>>>(xw, dinv, h, n128);
    k_agg_edges<<<gEC, B, 0, stream>>>(row, col, nrm, xw, h, E);
    k_zero_stats<<<1, 256, 0, stream>>>(stats);
    k_bn_stats<<<1024, 128, 0, stream>>>(h, b1, stats, N);
    k_bn_apply<<<gNC, B, 0, stream>>>(h, b1, g1, be1, stats, xb, N, n128);

    // ---- layer 2 ----
    k_w_to_bf16t<<<gW, B, 0, stream>>>(W2, wt, N_CH * N_CH);
    k_gemm_bf16<<<gT, B, 0, stream>>>(xb, wt, xw, N);
    k_agg_init<<<gNC, B, 0, stream>>>(xw, dinv, h, n128);
    k_agg_edges<<<gEC, B, 0, stream>>>(row, col, nrm, xw, h, E);
    k_zero_stats<<<1, 256, 0, stream>>>(stats);
    k_bn_stats<<<1024, 128, 0, stream>>>(h, b2, stats, N);
    k_bn_apply<<<gNC, B, 0, stream>>>(h, b2, g2, be2, stats, xb, N, n128);

    // ---- layer 3 (128 -> 1) ----
    k_matvec<<<gN, B, 0, stream>>>(xb, W3, xw, N);         // xw[0..N) = h2 @ W3
    k_out_init<<<gN, B, 0, stream>>>(xw, dinv, b3, out, N);
    k_out_edges<<<gE, B, 0, stream>>>(row, col, nrm, xw, out, E);
}
